// MyAttention_18021682774436
// MI455X (gfx1250) — matrix operand order repaired and checked by reading, not compiled, not run on hardware
//
#include <hip/hip_runtime.h>
#include <hip/hip_bf16.h>
#include <math.h>

// ---------------- problem constants ----------------
#define BDIM   2
#define NQ     4096
#define CDIM   512
#define NHEADS 8
#define HDIM   64
#define NSRC   2048
#define N0T    4096
#define HMAP   64
#define WMAP   64
#define M2     1024      // 32*32 downsampled tokens
#define CP1    513       // C + 1 (conf channel)
#define SCALE_ATT 0.125f // 64^-0.5

typedef __bf16 bf16_t;
typedef bf16_t v16bf __attribute__((ext_vector_type(16)));
typedef float  v8f   __attribute__((ext_vector_type(8)));

union VC { v8f v; float e[8]; };
union VB { v16bf v; uint4 q[2]; };

// ---------------- WMMA helpers (ISA 7.12.2 bf16 layouts) ----------------
__device__ __forceinline__ v8f wmma_bf16(v16bf a, v16bf b, v8f c) {
  return __builtin_amdgcn_wmma_f32_16x16x32_bf16(
      false, a, false, b, (short)0, c, false, false);
}

// A 16x32 (MxK) from row-major [M][K] LDS tile.
// lane&15 = row M; kbase = (lane>>4)*8; elems = K kbase..kbase+7, kbase+16..+23
// -> two contiguous 16B chunks => 2x ds_load_b128
__device__ __forceinline__ v16bf load_frag_A(const bf16_t* lds, int ld,
                                             int m0, int k0, int lane) {
  const bf16_t* p = lds + (size_t)(m0 + (lane & 15)) * ld
                        + k0 + ((lane >> 4) << 3);
  VB a;
  a.q[0] = *(const uint4*)(p);
  a.q[1] = *(const uint4*)(p + 16);
  return a.v;
}

// B 32x16 (KxN) from K-contiguous [N][K] LDS tile.
// lane&15 = col N; kbase = (lane>>4)*8; elems = K kbase..kbase+7, kbase+16..+23
// -> two contiguous 16B chunks => 2x ds_load_b128
__device__ __forceinline__ v16bf load_frag_Bt(const bf16_t* lds, int ld,
                                              int k0, int n0, int lane) {
  const bf16_t* p = lds + (size_t)(n0 + (lane & 15)) * ld
                        + k0 + ((lane >> 4) << 3);
  VB b;
  b.q[0] = *(const uint4*)(p);
  b.q[1] = *(const uint4*)(p + 16);
  return b.v;
}

// ---------------- weight conversion (plain cast, keep [N][K] layout) ---------
__global__ void castbf_k(const float* __restrict__ W, bf16_t* __restrict__ o, int n) {
  int i = blockIdx.x * blockDim.x + threadIdx.x;
  if (i < n) o[i] = (bf16_t)W[i];
}

// sr_w (oc,ic,ky,kx) -> srwB[oc][kk], kk = (ky*2+kx)*512 + ic  (K-contiguous)
__global__ void convertSRW_k(const float* __restrict__ W, bf16_t* __restrict__ o) {
  int j = blockIdx.x * blockDim.x + threadIdx.x;   // output index
  if (j >= CDIM * 2048) return;
  int oc = j >> 11;
  int rem = j & 2047;
  int q  = rem >> 9;      // ky*2+kx
  int ic = rem & 511;
  o[j] = (bf16_t)W[(size_t)oc * 2048 + ic * 4 + q];
}

// ---------------- token2map scatter ----------------
__global__ void scatter_k(const float* __restrict__ xsrc,
                          const float* __restrict__ conf,
                          const float* __restrict__ loc,
                          const int* __restrict__ idx,
                          float* __restrict__ map, float* __restrict__ cnt) {
  int token = blockIdx.x;          // 0 .. B*N0-1
  int b = token >> 12;             // /4096
  float lx = fminf(fmaxf(loc[(size_t)token * 2 + 0], -1.f), 1.f);
  float ly = fminf(fmaxf(loc[(size_t)token * 2 + 1], -1.f), 1.f);
  int px = (int)rintf(0.5f * (lx + 1.f) * WMAP - 0.5f);
  int py = (int)rintf(0.5f * (ly + 1.f) * HMAP - 0.5f);
  px = min(max(px, 0), WMAP - 1);
  py = min(max(py, 0), HMAP - 1);
  int pix = py * WMAP + px;
  int s = idx[token];
  const float* f = xsrc + ((size_t)b * NSRC + s) * CDIM;
  float* dst = map + ((size_t)b * (HMAP * WMAP) + pix) * CP1;
  for (int c = threadIdx.x; c < CDIM; c += blockDim.x)
    atomicAdd(&dst[c], f[c]);
  if (threadIdx.x == 0) {
    atomicAdd(&dst[CDIM], conf[(size_t)b * NSRC + s]);
    atomicAdd(&cnt[(size_t)b * (HMAP * WMAP) + pix], 1.0f);
  }
}

__global__ void finalize_k(float* __restrict__ map, const float* __restrict__ cnt) {
  size_t i = (size_t)blockIdx.x * blockDim.x + threadIdx.x;
  if (i >= (size_t)BDIM * HMAP * WMAP * CP1) return;
  size_t pix = i / CP1;
  map[i] = map[i] / (cnt[pix] + 1e-6f);
}

__global__ void confpool_k(const float* __restrict__ map, float* __restrict__ confp) {
  int i = blockIdx.x * blockDim.x + threadIdx.x;  // 0 .. B*M2-1
  if (i >= BDIM * M2) return;
  int b = i >> 10, p = i & 1023;
  int y2 = p >> 5, x2 = p & 31;
  float s = 0.f;
#pragma unroll
  for (int ky = 0; ky < 2; ++ky)
#pragma unroll
    for (int kx = 0; kx < 2; ++kx)
      s += map[((size_t)b * (HMAP * WMAP) + (y2 * 2 + ky) * WMAP + x2 * 2 + kx) * CP1 + CDIM];
  confp[i] = s * 0.25f;
}

// ---------------- generic WMMA GEMM: C[M][N] = A[M][K] * W[N][K]^T (+bias) -----
// A from f32 (Af) or bf16 (Ab). W is bf16 [N][K] (natural layout = B^T).
// 256 threads = 8 waves; block tile 64(M) x 64(N); each wave: 1 A frag x 2 B frags.
__global__ __launch_bounds__(256)
void gemm_tn(const float* __restrict__ Af, const bf16_t* __restrict__ Ab,
             const bf16_t* __restrict__ Wb, const float* __restrict__ bias,
             float* __restrict__ Cf, bf16_t* __restrict__ Cb,
             int M, int N, int K) {
  __shared__ __align__(16) bf16_t As[64][40];   // [m][k]
  __shared__ __align__(16) bf16_t Bs[64][40];   // [n][k]  (K-contiguous)
  const int tid = threadIdx.x, lane = tid & 31, wave = tid >> 5;
  const int m_t = (wave & 3) * 16;
  const int nb  = (wave >> 2) * 32;
  const int bm = blockIdx.x * 64, bn = blockIdx.y * 64;
  const int ar = tid >> 2, ac = (tid & 3) * 8;   // 8 contiguous per thread
  VC acc0, acc1;
#pragma unroll
  for (int r = 0; r < 8; ++r) { acc0.e[r] = 0.f; acc1.e[r] = 0.f; }

  for (int kt = 0; kt < K; kt += 32) {
    if (Af) {
      const float* src = Af + (size_t)(bm + ar) * K + kt + ac;
      if (kt + 32 < K) __builtin_prefetch(src + 32, 0, 0);
#pragma unroll
      for (int j = 0; j < 8; ++j) As[ar][ac + j] = (bf16_t)src[j];
    } else {
      const bf16_t* src = Ab + (size_t)(bm + ar) * K + kt + ac;
      if (kt + 32 < K) __builtin_prefetch(src + 32, 0, 0);
#pragma unroll
      for (int j = 0; j < 8; ++j) As[ar][ac + j] = src[j];
    }
    {
      const bf16_t* bsrc = Wb + (size_t)(bn + ar) * K + kt + ac;
#pragma unroll
      for (int j = 0; j < 8; ++j) Bs[ar][ac + j] = bsrc[j];
    }
    __syncthreads();
    v16bf a  = load_frag_A (&As[0][0], 40, m_t, 0, lane);
    v16bf b0 = load_frag_Bt(&Bs[0][0], 40, 0, nb, lane);
    v16bf b1 = load_frag_Bt(&Bs[0][0], 40, 0, nb + 16, lane);
    acc0.v = wmma_bf16(a, b0, acc0.v);
    acc1.v = wmma_bf16(a, b1, acc1.v);
    __syncthreads();
  }
  const int rowbase = bm + m_t + ((lane < 16) ? 0 : 8);
  const int col0 = bn + nb + (lane & 15);
  const float bv0 = bias ? bias[col0] : 0.f;
  const float bv1 = bias ? bias[col0 + 16] : 0.f;
#pragma unroll
  for (int r = 0; r < 8; ++r) {
    size_t o = (size_t)(rowbase + r) * N + col0;
    float v0 = acc0.e[r] + bv0, v1 = acc1.e[r] + bv1;
    if (Cf) { Cf[o] = v0; Cf[o + 16] = v1; }
    else    { Cb[o] = (bf16_t)v0; Cb[o + 16] = (bf16_t)v1; }
  }
}

// ---------------- conv 2x2 stride-2 as WMMA GEMM (im2col on the fly) -----------
// M = B*1024 (row = b*1024 + y*32 + x), N = 512, K = 2048; W = srwB[oc][kk]
__global__ __launch_bounds__(256)
void conv_gemm(const float* __restrict__ map, const bf16_t* __restrict__ srwB,
               const float* __restrict__ srb, float* __restrict__ xs) {
  __shared__ __align__(16) bf16_t As[64][40];
  __shared__ __align__(16) bf16_t Bs[64][40];
  const int tid = threadIdx.x, lane = tid & 31, wave = tid >> 5;
  const int m_t = (wave & 3) * 16;
  const int nb  = (wave >> 2) * 32;
  const int bm = blockIdx.x * 64, bn = blockIdx.y * 64;
  const int ar = tid >> 2, ac = (tid & 3) * 8;
  const int grow = bm + ar;
  const int b = grow >> 10, p = grow & 1023;
  const int y = p >> 5, x = p & 31;
  VC acc0, acc1;
#pragma unroll
  for (int r = 0; r < 8; ++r) { acc0.e[r] = 0.f; acc1.e[r] = 0.f; }

  for (int kt = 0; kt < 2048; kt += 32) {
    int seg = kt >> 9;             // which (ky,kx) block (512 | 32)
    int ky = seg >> 1, kx = seg & 1;
    int ic0 = kt & 511;
    int pix = ((y * 2 + ky) << 6) + x * 2 + kx;
    const float* src = map + ((size_t)(b * (HMAP * WMAP) + pix)) * CP1 + ic0 + ac;
#pragma unroll
    for (int j = 0; j < 8; ++j) As[ar][ac + j] = (bf16_t)src[j];
    const bf16_t* bsrc = srwB + (size_t)(bn + ar) * 2048 + kt + ac;
#pragma unroll
    for (int j = 0; j < 8; ++j) Bs[ar][ac + j] = bsrc[j];
    __syncthreads();
    v16bf a  = load_frag_A (&As[0][0], 40, m_t, 0, lane);
    v16bf b0 = load_frag_Bt(&Bs[0][0], 40, 0, nb, lane);
    v16bf b1 = load_frag_Bt(&Bs[0][0], 40, 0, nb + 16, lane);
    acc0.v = wmma_bf16(a, b0, acc0.v);
    acc1.v = wmma_bf16(a, b1, acc1.v);
    __syncthreads();
  }
  const int rowbase = bm + m_t + ((lane < 16) ? 0 : 8);
  const int col0 = bn + nb + (lane & 15);
  const float bv0 = srb[col0], bv1 = srb[col0 + 16];
#pragma unroll
  for (int r = 0; r < 8; ++r) {
    size_t o = (size_t)(rowbase + r) * CDIM + col0;
    xs[o] = acc0.e[r] + bv0;
    xs[o + 16] = acc1.e[r] + bv1;
  }
}

// ---------------- layernorm over C, output bf16 ----------------
__global__ __launch_bounds__(256)
void layernorm_k(const float* __restrict__ xs, const float* __restrict__ g,
                 const float* __restrict__ bta, bf16_t* __restrict__ out) {
  __shared__ float red[256];
  const int row = blockIdx.x;
  const float* p = xs + (size_t)row * CDIM;
  float s = 0.f, s2 = 0.f;
  for (int c = threadIdx.x; c < CDIM; c += 256) { float v = p[c]; s += v; s2 += v * v; }
  red[threadIdx.x] = s; __syncthreads();
  for (int o = 128; o > 0; o >>= 1) { if (threadIdx.x < o) red[threadIdx.x] += red[threadIdx.x + o]; __syncthreads(); }
  float mean = red[0] / (float)CDIM; __syncthreads();
  red[threadIdx.x] = s2; __syncthreads();
  for (int o = 128; o > 0; o >>= 1) { if (threadIdx.x < o) red[threadIdx.x] += red[threadIdx.x + o]; __syncthreads(); }
  float var = red[0] / (float)CDIM - mean * mean;
  float rstd = rsqrtf(var + 1e-5f);
  for (int c = threadIdx.x; c < CDIM; c += 256)
    out[(size_t)row * CDIM + c] = (bf16_t)((p[c] - mean) * rstd * g[c] + bta[c]);
}

// ---------------- flash attention: 64-query tile per (b,h) block ----------------
// kv packed as [B*M2][1024]: cols 0..511 = k, 512..1023 = v (head-major 64-wide)
__global__ __launch_bounds__(256)
void attn_k(const bf16_t* __restrict__ qb, const bf16_t* __restrict__ kvb,
            const float* __restrict__ confp, float* __restrict__ out) {
  __shared__ __align__(16) bf16_t Qs[64][72];   // [query][dim]
  __shared__ __align__(16) bf16_t Ks[32][72];   // [key][dim]   (B^T for S)
  __shared__ __align__(16) bf16_t Vst[64][40];  // [dim][key]   (B^T for PV)
  __shared__ __align__(16) bf16_t Ps[64][40];   // [query][key] (A for PV)
  __shared__ float  Ss[64][34];
  __shared__ float  Os[64][66];
  __shared__ float  mrow[64], lrow[64], frow[64];

  const int tid = threadIdx.x, lane = tid & 31, wave = tid >> 5;
  const int qt = blockIdx.x & 63;
  const int h  = (blockIdx.x >> 6) & 7;
  const int b  = blockIdx.x >> 9;
  const int n0 = qt * 64;
  const int m_t = (wave >> 1) * 16;
  const int n_t16 = (wave & 1) * 16;

  {
    const bf16_t* src = qb + ((size_t)(b * NQ + n0)) * CDIM + h * HDIM;
    for (int e = tid; e < 64 * 64; e += 256)
      Qs[e >> 6][e & 63] = src[(size_t)(e >> 6) * CDIM + (e & 63)];
  }
  for (int e = tid; e < 64 * 64; e += 256) Os[e >> 6][e & 63] = 0.f;
  if (tid < 64) { mrow[tid] = -1e30f; lrow[tid] = 0.f; }
  __syncthreads();

  for (int kc = 0; kc < M2; kc += 32) {
    for (int e = tid; e < 32 * 64; e += 256) {
      int r = e >> 6, c = e & 63;           // r = key, c = dim
      size_t base = ((size_t)(b * M2 + kc + r)) * 1024 + h * HDIM + c;
      Ks[r][c]  = kvb[base];
      Vst[c][r] = kvb[base + CDIM];         // store V transposed
    }
    __syncthreads();

    // S = Q(64x64) * K^T(64x32): each wave one 16x16 tile, 2 K-steps
    VC acc;
#pragma unroll
    for (int r = 0; r < 8; ++r) acc.e[r] = 0.f;
#pragma unroll
    for (int ks = 0; ks < 64; ks += 32) {
      v16bf a  = load_frag_A (&Qs[0][0], 72, m_t, ks, lane);
      v16bf bb = load_frag_Bt(&Ks[0][0], 72, ks, n_t16, lane);
      acc.v = wmma_bf16(a, bb, acc.v);
    }
    {
      int rb = m_t + ((lane < 16) ? 0 : 8);
      int col = n_t16 + (lane & 15);
      float cb = confp[b * M2 + kc + col];
#pragma unroll
      for (int r = 0; r < 8; ++r)
        Ss[rb + r][col] = acc.e[r] * SCALE_ATT + cb;
    }
    __syncthreads();

    if (tid < 64) {                         // online softmax row stats
      float mo = mrow[tid], mc = mo;
#pragma unroll
      for (int j = 0; j < 32; ++j) mc = fmaxf(mc, Ss[tid][j]);
      float f = __expf(mo - mc);
      float s = 0.f;
#pragma unroll
      for (int j = 0; j < 32; ++j) {
        float pv = __expf(Ss[tid][j] - mc);
        Ps[tid][j] = (bf16_t)pv;
        s += pv;
      }
      mrow[tid] = mc;
      lrow[tid] = lrow[tid] * f + s;
      frow[tid] = f;
    }
    __syncthreads();
    for (int e = tid; e < 64 * 64; e += 256) Os[e >> 6][e & 63] *= frow[e >> 6];
    __syncthreads();

    // O += P(64x32) * V(32x64): each wave two 16x16 tiles
    {
      v16bf a = load_frag_A(&Ps[0][0], 40, m_t, 0, lane);
      int nbv = (wave & 1) * 32;
#pragma unroll
      for (int t = 0; t < 2; ++t) {
        int nt = nbv + t * 16;
        v16bf bb = load_frag_Bt(&Vst[0][0], 40, 0, nt, lane);
        VC z;
#pragma unroll
        for (int r = 0; r < 8; ++r) z.e[r] = 0.f;
        z.v = wmma_bf16(a, bb, z.v);
        int rb = m_t + ((lane < 16) ? 0 : 8);
        int col = nt + (lane & 15);
#pragma unroll
        for (int r = 0; r < 8; ++r) Os[rb + r][col] += z.e[r];
      }
    }
    __syncthreads();
  }

  for (int e = tid; e < 64 * 64; e += 256) {
    int r = e >> 6, c = e & 63;
    out[((size_t)(b * NQ + n0 + r)) * CDIM + h * HDIM + c] = Os[r][c] / lrow[r];
  }
}

// ---------------- workspace layout (bytes) ----------------
constexpr size_t MAP_BYTES = (size_t)BDIM * HMAP * WMAP * CP1 * 4; // reused as attn out
constexpr size_t OFF_MAP  = 0;
constexpr size_t OFF_CNT  = OFF_MAP  + MAP_BYTES;
constexpr size_t OFF_XS   = OFF_CNT  + (size_t)BDIM * HMAP * WMAP * 4;
constexpr size_t OFF_CONF = OFF_XS   + (size_t)BDIM * M2 * CDIM * 4;
constexpr size_t OFF_Q    = OFF_CONF + (size_t)BDIM * M2 * 4;
constexpr size_t OFF_KV   = OFF_Q    + (size_t)BDIM * NQ * CDIM * 2;
constexpr size_t OFF_XSB  = OFF_KV   + (size_t)BDIM * M2 * 1024 * 2;
constexpr size_t OFF_WQB  = OFF_XSB  + (size_t)BDIM * M2 * CDIM * 2;
constexpr size_t OFF_WKVB = OFF_WQB  + (size_t)CDIM * CDIM * 2;
constexpr size_t OFF_SRWB = OFF_WKVB + (size_t)1024 * CDIM * 2;
constexpr size_t OFF_PRJB = OFF_SRWB + (size_t)2048 * CDIM * 2;

extern "C" void kernel_launch(void* const* d_in, const int* in_sizes, int n_in,
                              void* d_out, int out_size, void* d_ws, size_t ws_size,
                              hipStream_t stream) {
  const float* x     = (const float*)d_in[0];
  const float* loc   = (const float*)d_in[1];
  const float* xsrc  = (const float*)d_in[2];
  const int*   idx   = (const int*)  d_in[3];
  const float* confs = (const float*)d_in[6];
  const float* Wq    = (const float*)d_in[7];
  const float* Wkv   = (const float*)d_in[8];
  const float* srw   = (const float*)d_in[9];
  const float* srb   = (const float*)d_in[10];
  const float* lng   = (const float*)d_in[11];
  const float* lnb   = (const float*)d_in[12];
  const float* prjw  = (const float*)d_in[13];
  const float* prjb  = (const float*)d_in[14];
  float* out = (float*)d_out;

  char* ws = (char*)d_ws;
  float*  mapb  = (float*)(ws + OFF_MAP);
  float*  cntb  = (float*)(ws + OFF_CNT);
  float*  xsf   = (float*)(ws + OFF_XS);
  float*  confp = (float*)(ws + OFF_CONF);
  bf16_t* qB    = (bf16_t*)(ws + OFF_Q);
  bf16_t* kvB   = (bf16_t*)(ws + OFF_KV);
  bf16_t* xsB   = (bf16_t*)(ws + OFF_XSB);
  bf16_t* wqB   = (bf16_t*)(ws + OFF_WQB);
  bf16_t* wkvB  = (bf16_t*)(ws + OFF_WKVB);
  bf16_t* srwB  = (bf16_t*)(ws + OFF_SRWB);
  bf16_t* prjB  = (bf16_t*)(ws + OFF_PRJB);
  float*  attn_out = mapb;   // reuse map region after conv consumes it

  hipMemsetAsync(mapb, 0, MAP_BYTES + (size_t)BDIM * HMAP * WMAP * 4, stream);

  castbf_k<<<(512 * 512 + 255) / 256, 256, 0, stream>>>(Wq, wqB, 512 * 512);
  castbf_k<<<(1024 * 512 + 255) / 256, 256, 0, stream>>>(Wkv, wkvB, 1024 * 512);
  castbf_k<<<(512 * 512 + 255) / 256, 256, 0, stream>>>(prjw, prjB, 512 * 512);
  convertSRW_k<<<(512 * 2048 + 255) / 256, 256, 0, stream>>>(srw, srwB);

  scatter_k<<<BDIM * N0T, 128, 0, stream>>>(xsrc, confs, loc, idx, mapb, cntb);
  finalize_k<<<(BDIM * HMAP * WMAP * CP1 + 255) / 256, 256, 0, stream>>>(mapb, cntb);
  confpool_k<<<(BDIM * M2 + 255) / 256, 256, 0, stream>>>(mapb, confp);

  conv_gemm<<<dim3(2048 / 64, 512 / 64), 256, 0, stream>>>(mapb, srwB, srb, xsf);
  layernorm_k<<<BDIM * M2, 256, 0, stream>>>(xsf, lng, lnb, xsB);

  // q = x @ Wq^T  (M=8192, N=512, K=512) -> bf16
  gemm_tn<<<dim3(8192 / 64, 512 / 64), 256, 0, stream>>>(
      x, nullptr, wqB, nullptr, nullptr, qB, 8192, 512, 512);
  // kv = xs @ Wkv^T (M=2048, N=1024, K=512) -> bf16 packed k|v
  gemm_tn<<<dim3(2048 / 64, 1024 / 64), 256, 0, stream>>>(
      nullptr, xsB, wkvB, nullptr, nullptr, kvB, 2048, 1024, 512);

  attn_k<<<BDIM * NHEADS * (NQ / 64), 256, 0, stream>>>(qB, kvB, confp, attn_out);

  // out = attn_out @ proj_w^T + proj_b (M=8192, N=512, K=512) -> f32
  gemm_tn<<<dim3(8192 / 64, 512 / 64), 256, 0, stream>>>(
      attn_out, nullptr, prjB, prjb, out, nullptr, 8192, 512, 512);
}
